// Net_67456756351352
// MI455X (gfx1250) — compile-verified, matrix operand order checked by reading, not run on hardware
//
#include <hip/hip_runtime.h>
#include <hip/hip_bf16.h>
#include <math.h>

// ---------------------------------------------------------------------------
// GCN link predictor for MI455X (gfx1250, wave32).
//   - GEMMs via v_wmma_f32_16x16x32_bf16, fragments pre-packed per lane so
//     A comes from LDS as 2x ds_load_b128 and B from global as 2x
//     global_load_b128 per chunk, k-loop fully unrolled (4 WMMAs/tile).
//   - Message passing via thread-per-(edge,feature) global_atomic_add_f32;
//     agg arrays (20MB) are L2-resident on the 192MB L2.
// ---------------------------------------------------------------------------

typedef __attribute__((ext_vector_type(16))) __bf16 v16bf;
typedef __attribute__((ext_vector_type(8)))  float  v8f;

#define KPAD 128          // padded K for both GEMMs (128 and 100->128)
#define KCH  4            // KPAD / 32 k-chunks

// ---- weight pre-pack: f32 [K][N] -> per-lane WMMA B fragments -------------
// layout: out[((nt*KCH + kc)*32 + lane)*16 + e]  (bf16)
__global__ void pack_w_bf16(const float* __restrict__ W, __bf16* __restrict__ out,
                            int K, int N, int NT) {
  int tid = blockIdx.x * blockDim.x + threadIdx.x;   // (nt*KCH+kc)*32 + lane
  int total = NT * KCH * 32;
  if (tid >= total) return;
  int lane = tid & 31;
  int kc   = (tid >> 5) & (KCH - 1);
  int nt   = tid >> 7;
  int n    = nt * 16 + (lane & 15);
  int kh   = (lane >> 4) << 3;
#pragma unroll
  for (int e = 0; e < 16; ++e) {
    int vg = e >> 1, pos = e & 1;
    int kl = ((vg & 3) << 1) + ((vg >> 2) << 4) + kh + pos;  // ISA 7.12.2
    int k  = (kc << 5) + kl;
    float v = (k < K && n < N) ? W[(size_t)k * N + n] : 0.0f;
    out[(size_t)tid * 16 + e] = (__bf16)v;
  }
}

// ---- degree / norm --------------------------------------------------------
__global__ void init_deg(float* __restrict__ deg, int N) {
  int n = blockIdx.x * blockDim.x + threadIdx.x;
  if (n < N) deg[n] = 1.0f;                       // self-loop
}

__global__ void deg_scatter(const int* __restrict__ dst, float* __restrict__ deg, int E) {
  int e = blockIdx.x * blockDim.x + threadIdx.x;
  if (e < E) atomicAdd(&deg[dst[e]], 1.0f);
}

__global__ void rsqrt_deg(const float* __restrict__ deg, float* __restrict__ dis, int N) {
  int n = blockIdx.x * blockDim.x + threadIdx.x;
  if (n < N) {
    float d = deg[n];
    dis[n] = d > 0.0f ? rsqrtf(d) : 0.0f;
  }
}

// ---- WMMA GEMM: C[M][N] = A[M][K] (f32 row major) x Bp (packed fragments) -
// grid.x = M/16 ; blockDim.x = ntiles*32 ; wave w computes n-tile w.
__global__ void wmma_gemm_bf16(const float* __restrict__ A,
                               const v16bf* __restrict__ Bp,
                               float* __restrict__ C, int M, int K, int N) {
  // A tile staged directly in fragment-packed order: [kc][lane][16 elems]
  __shared__ __attribute__((aligned(16))) __bf16 As[KCH * 32 * 16];
  const int mbase = blockIdx.x * 16;

  for (int idx = threadIdx.x; idx < 16 * KPAD; idx += blockDim.x) {
    int m = idx >> 7;              // row in tile
    int k = idx & (KPAD - 1);
    float v = (k < K) ? A[(size_t)(mbase + m) * K + k] : 0.0f;
    // inverse of kl = 2*(vg&3) + 16*(vg>>2) + 8*half + pos
    int kc   = k >> 5;
    int kl   = k & 31;
    int half = (kl >> 3) & 1;
    int klg  = kl - (half << 3);   // {0..7} u {16..23}
    int vg   = (klg < 8) ? (klg >> 1) : (4 + ((klg - 16) >> 1));
    int e    = (vg << 1) | (klg & 1);
    As[(((kc << 5) + m + (half << 4)) << 4) + e] = (__bf16)v;
  }
  __syncthreads();

  const int w    = threadIdx.x >> 5;       // wave == n-tile
  const int lane = threadIdx.x & 31;
  const v16bf* Asp = (const v16bf*)As;

  v8f c = {};
#pragma unroll
  for (int kc = 0; kc < KCH; ++kc) {
    v16bf a = Asp[(kc << 5) + lane];                     // 2x ds_load_b128
    v16bf b = Bp[(size_t)((w << 2) + kc) * 32 + lane];   // 2x global_load_b128
    c = __builtin_amdgcn_wmma_f32_16x16x32_bf16(false, a, false, b,
                                                (short)0, c, false, false);
  }

  const int col  = (w << 4) + (lane & 15); // C/D: lane col = lane%16
  const int rowh = mbase + ((lane >> 4) << 3);
  if (col < N) {
    float* cp = C + (size_t)rowh * N + col;
#pragma unroll
    for (int r = 0; r < 8; ++r) cp[(size_t)r * N] = c[r];
  }
}

// ---- aggregation ----------------------------------------------------------
// init agg with the self-loop contribution: agg[n][f] = h[n][f] * dis[n]^2
__global__ void init_agg(const float* __restrict__ h, const float* __restrict__ dis,
                         float* __restrict__ agg, int total, int F) {
  int tid = blockIdx.x * blockDim.x + threadIdx.x;
  if (tid >= total) return;
  int n = tid / F;
  float d = dis[n];
  agg[tid] = h[tid] * d * d;
}

// agg[dst][f] += h[src][f] * dis[src]*dis[dst], thread per (edge, feature)
__global__ void edge_scatter(const int* __restrict__ src, const int* __restrict__ dst,
                             const float* __restrict__ dis, const float* __restrict__ h,
                             float* __restrict__ agg, long long total, int F) {
  long long tid = (long long)blockIdx.x * blockDim.x + threadIdx.x;
  if (tid >= total) return;
  int e = (int)(tid / F);
  int f = (int)(tid % F);
  int s = src[e], d = dst[e];
  float nrm = dis[s] * dis[d];
  atomicAdd(&agg[(size_t)d * F + f], h[(size_t)s * F + f] * nrm);
}

__global__ void bias_relu(float* __restrict__ agg, const float* __restrict__ b,
                          int total, int F) {
  int tid = blockIdx.x * blockDim.x + threadIdx.x;
  if (tid >= total) return;
  float v = agg[tid] + b[tid % F];
  agg[tid] = v > 0.0f ? v : 0.0f;
}

// ---- row clip-normalize (H2 == 16, float4 x4) -----------------------------
__global__ void clip_normalize(float* __restrict__ emb, int N) {
  int n = blockIdx.x * blockDim.x + threadIdx.x;
  if (n >= N) return;
  float4* row = (float4*)(emb + (size_t)n * 16);
  float4 v[4];
  float ss = 0.0f;
#pragma unroll
  for (int i = 0; i < 4; ++i) {
    v[i] = row[i];
    ss += v[i].x * v[i].x + v[i].y * v[i].y + v[i].z * v[i].z + v[i].w * v[i].w;
  }
  float nrm = sqrtf(ss);
  float sc  = nrm > 1.0f ? 1.0f / (nrm + 1e-7f) : 1.0f;
#pragma unroll
  for (int i = 0; i < 4; ++i) {
    v[i].x *= sc; v[i].y *= sc; v[i].z *= sc; v[i].w *= sc;
    row[i] = v[i];
  }
}

// ---- edge decoder ---------------------------------------------------------
// feat = [(emb_in-emb_out)^2 (16), PI (25)]; hdec = leaky(feat@Wl1+bl1, 0.2);
// s = clip(|hdec@Wl + bl|, 0, 40); prob = 1/(exp(s-2)+1)
__global__ void decoder_kernel(const float* __restrict__ emb,
                               const int*   __restrict__ te,
                               const float* __restrict__ PI,
                               const float* __restrict__ Wl1,
                               const float* __restrict__ bl1,
                               const float* __restrict__ Wl,
                               const float* __restrict__ bl,
                               const float* __restrict__ edges_y,
                               float* __restrict__ out, int ED) {
  const int H2 = 16, D2 = 25, FEAT = 41;
  __shared__ float sW[FEAT * D2 + D2 + D2 + 1];   // Wl1 | bl1 | Wl | bl
  for (int i = threadIdx.x; i < FEAT * D2 + 2 * D2 + 1; i += blockDim.x) {
    float v;
    if      (i < FEAT * D2)          v = Wl1[i];
    else if (i < FEAT * D2 + D2)     v = bl1[i - FEAT * D2];
    else if (i < FEAT * D2 + 2 * D2) v = Wl[i - FEAT * D2 - D2];
    else                             v = bl[0];
    sW[i] = v;
  }
  __syncthreads();
  const float* sWl1 = sW;
  const float* sbl1 = sW + FEAT * D2;
  const float* sWl  = sW + FEAT * D2 + D2;
  const float  sbl  = sW[FEAT * D2 + 2 * D2];

  int e = blockIdx.x * blockDim.x + threadIdx.x;
  if (e >= ED) return;

  int ia = te[2 * e], ib = te[2 * e + 1];
  const float4* ra = (const float4*)(emb + (size_t)ia * H2);
  const float4* rb = (const float4*)(emb + (size_t)ib * H2);

  float feat[FEAT];
#pragma unroll
  for (int i = 0; i < 4; ++i) {
    float4 a = ra[i], b = rb[i];
    float dx = a.x - b.x, dy = a.y - b.y, dz = a.z - b.z, dw = a.w - b.w;
    feat[4 * i + 0] = dx * dx; feat[4 * i + 1] = dy * dy;
    feat[4 * i + 2] = dz * dz; feat[4 * i + 3] = dw * dw;
  }
#pragma unroll
  for (int j = 0; j < D2; ++j) feat[H2 + j] = PI[(size_t)e * D2 + j];

  float acc = sbl;
#pragma unroll 1
  for (int j = 0; j < D2; ++j) {
    float v = sbl1[j];
    for (int i = 0; i < FEAT; ++i) v = fmaf(feat[i], sWl1[i * D2 + j], v);
    v = v > 0.0f ? v : 0.2f * v;                 // leaky_relu
    acc = fmaf(v, sWl[j], acc);
  }
  float s = fabsf(acc);
  s = fminf(fmaxf(s, 0.0f), 40.0f);
  out[e]      = 1.0f / (expf(s - 2.0f) + 1.0f);
  out[ED + e] = edges_y[e];
}

// ---------------------------------------------------------------------------
extern "C" void kernel_launch(void* const* d_in, const int* in_sizes, int n_in,
                              void* d_out, int out_size, void* d_ws, size_t ws_size,
                              hipStream_t stream) {
  const float* x    = (const float*)d_in[0];
  const int*   ei   = (const int*)d_in[1];    // edge_index (2,E) int32
  const int*   te   = (const int*)d_in[2];    // total_edges (ED,2) int32
  const float* PI   = (const float*)d_in[3];
  const float* ey   = (const float*)d_in[4];
  const float* W1   = (const float*)d_in[5];
  const float* b1   = (const float*)d_in[6];
  const float* W2   = (const float*)d_in[7];
  const float* b2   = (const float*)d_in[8];
  const float* Wl1  = (const float*)d_in[9];
  const float* bl1  = (const float*)d_in[10];
  const float* Wl   = (const float*)d_in[11];
  const float* bl   = (const float*)d_in[12];
  float* out = (float*)d_out;

  const int F  = 128;
  const int H1 = in_sizes[6];                 // 100
  const int H2 = in_sizes[8];                 // 16
  const int Nn = in_sizes[0] / F;             // 50000
  const int E  = in_sizes[1] / 2;             // 800000
  const int ED = in_sizes[4];                 // 200000
  const int NT1 = (H1 + 15) / 16;             // 7
  const int NT2 = (H2 + 15) / 16;             // 1

  // workspace carve (256B aligned)
  char* base = (char*)d_ws;
  size_t off = 0;
  auto carve = [&](size_t bytes) {
    char* p = base + off;
    off = (off + bytes + 255) & ~(size_t)255;
    return p;
  };
  float*  deg   = (float*)carve((size_t)Nn * 4);
  float*  dis   = (float*)carve((size_t)Nn * 4);
  float*  h1pre = (float*)carve((size_t)Nn * H1 * 4);
  float*  agg1  = (float*)carve((size_t)Nn * H1 * 4);   // becomes h1 in place
  float*  h2pre = (float*)carve((size_t)Nn * H2 * 4);
  float*  agg2  = (float*)carve((size_t)Nn * H2 * 4);   // becomes emb in place
  __bf16* W1p   = (__bf16*)carve((size_t)NT1 * KCH * 32 * 16 * 2);
  __bf16* W2p   = (__bf16*)carve((size_t)NT2 * KCH * 32 * 16 * 2);
  (void)ws_size;

  const int* src = ei;
  const int* dst = ei + E;
  const int TB = 256;
  auto blocks = [&](long long n) { return (unsigned)((n + TB - 1) / TB); };

  // weights -> packed per-lane bf16 fragments
  pack_w_bf16<<<blocks(NT1 * KCH * 32), TB, 0, stream>>>(W1, W1p, F,  H1, NT1);
  pack_w_bf16<<<blocks(NT2 * KCH * 32), TB, 0, stream>>>(W2, W2p, H1, H2, NT2);

  // GCN symmetric normalization
  init_deg   <<<blocks(Nn), TB, 0, stream>>>(deg, Nn);
  deg_scatter<<<blocks(E),  TB, 0, stream>>>(dst, deg, E);
  rsqrt_deg  <<<blocks(Nn), TB, 0, stream>>>(deg, dis, Nn);

  // layer 1: h1pre = x @ W1 (WMMA bf16), aggregate, +b1, relu
  wmma_gemm_bf16<<<Nn / 16, NT1 * 32, 0, stream>>>(x, (const v16bf*)W1p, h1pre,
                                                   Nn, F, H1);
  init_agg   <<<blocks((long long)Nn * H1), TB, 0, stream>>>(h1pre, dis, agg1, Nn * H1, H1);
  edge_scatter<<<blocks((long long)E * H1), TB, 0, stream>>>(src, dst, dis, h1pre, agg1,
                                                             (long long)E * H1, H1);
  bias_relu  <<<blocks((long long)Nn * H1), TB, 0, stream>>>(agg1, b1, Nn * H1, H1);

  // layer 2: h2pre = h1 @ W2 (WMMA bf16, K padded 100->128), aggregate, +b2, relu
  wmma_gemm_bf16<<<Nn / 16, NT2 * 32, 0, stream>>>(agg1, (const v16bf*)W2p, h2pre,
                                                   Nn, H1, H2);
  init_agg   <<<blocks((long long)Nn * H2), TB, 0, stream>>>(h2pre, dis, agg2, Nn * H2, H2);
  edge_scatter<<<blocks((long long)E * H2), TB, 0, stream>>>(src, dst, dis, h2pre, agg2,
                                                             (long long)E * H2, H2);
  bias_relu  <<<blocks((long long)Nn * H2), TB, 0, stream>>>(agg2, b2, Nn * H2, H2);

  // clip-normalize rows -> emb (in place in agg2)
  clip_normalize<<<blocks(Nn), TB, 0, stream>>>(agg2, Nn);

  // decoder: prob -> out[0:ED], edges_y copy -> out[ED:2*ED]
  decoder_kernel<<<blocks(ED), TB, 0, stream>>>(agg2, te, PI, Wl1, bl1, Wl, bl,
                                                ey, out, ED);
}
